// SlidingWindowAttention_7421703488075
// MI455X (gfx1250) — matrix operand order repaired and checked by reading, not compiled, not run on hardware
//
#include <hip/hip_runtime.h>
#include <hip/hip_bf16.h>

typedef __attribute__((ext_vector_type(16))) _Float16 v16h;
typedef __attribute__((ext_vector_type(8)))  _Float16 v8h;
typedef __attribute__((ext_vector_type(8)))  float    v8f;

#define S_LEN 4096
#define DMODEL 1024
#define NHEAD 16
#define HDIM 64
#define WIN 512

union FragU { v16h v; struct { v8h lo; v8h hi; } p; };

// A-fragment (16x32 f16, MxK): lane holds row; K chunks at col+hi*8 and col+16+hi*8
__device__ inline v16h load_frag_a(const _Float16* row_ptr, int colbase, int hi) {
  FragU u;
  u.p.lo = *(const v8h*)(row_ptr + colbase + hi * 8);
  u.p.hi = *(const v8h*)(row_ptr + colbase + 16 + hi * 8);
  return u.v;
}

// B-fragment (32x16 f16, KxN): lane holds column N (= a row of a transposed-stored
// matrix); K chunks at col+hi*8 and col+16+hi*8
__device__ inline v16h load_frag_b(const _Float16* row_ptr, int colbase, int hi) {
  FragU u;
  const _Float16* p = row_ptr + colbase + hi * 8;
  u.p.lo = *(const v8h*)(p);
  u.p.hi = *(const v8h*)(p + 16);
  return u.v;
}

__device__ inline v8f wmma_f16(v16h a, v16h b, v8f c) {
  return __builtin_amdgcn_wmma_f32_16x16x32_f16(false, a, false, b, (short)0, c, false, false);
}

// gfx1250 async DMA: one b128 (16B) per lane, global -> LDS, tracked by ASYNCcnt.
__device__ inline void async_copy_b128(uint32_t lds_off, const void* gptr) {
  asm volatile("global_load_async_to_lds_b128 %0, %1, off"
               :: "v"(lds_off), "v"((uint64_t)(uintptr_t)gptr) : "memory");
}

// ---------------- converts ----------------

__global__ __launch_bounds__(256) void cvt_f32_to_f16(const float* __restrict__ in,
                                                      _Float16* __restrict__ out, int n) {
  int i = blockIdx.x * 256 + threadIdx.x;
  if (i < n) out[i] = (_Float16)in[i];
}

// W [K=1024][N=1024] row-major  ->  Wt[n][k] f16
__global__ __launch_bounds__(256) void cvt_transpose(const float* __restrict__ in,
                                                     _Float16* __restrict__ out) {
  int i = blockIdx.x * 256 + threadIdx.x;      // over 1M elements
  int k = i >> 10, n = i & (DMODEL - 1);
  out[n * DMODEL + k] = (_Float16)in[i];
}

// ---------------- WMMA GEMM: out = A[4096x1024] * Wt^T + bias ----------------
// A (128x64) and B (64x64) k-step tiles staged in LDS via async DMA, double
// buffered; 8 WMMAs per barrier pair; fragment loads batched ahead of the
// WMMA chain so DS latency overlaps the matrix pipe.
// mode 0/1: store f16 to [H][S][64] (Q / K)
// mode 2  : store f16 to [H][64][S] (V transposed)
// mode 3  : store f32 to d_out [S][1024]
__global__ __launch_bounds__(256) void wmma_gemm_kernel(
    const _Float16* __restrict__ A,
    const _Float16* __restrict__ Bt,     // Bt[n][k]
    const float* __restrict__ bias,
    _Float16* __restrict__ out16,
    float* __restrict__ out32,
    int mode) {
  __shared__ __align__(16) _Float16 ash[2][128][64];  // 2 x 16KB
  __shared__ __align__(16) _Float16 bsh[2][64][64];   // 2 x 8KB

  const int tid  = threadIdx.x;
  const int lane = tid & 31;
  const int wave = tid >> 5;
  const int hi = lane >> 4;
  const int ln = lane & 15;
  const int mblk = blockIdx.x * 128;
  const int m0 = mblk + wave * 16;
  const int n0 = blockIdx.y * 64;

  // DMA mapping (16B chunks) per 64-k step: B = 512 chunks (2/thr), A = 1024 (4/thr)
  auto prefetch = [&](int b, int k0) {
#pragma unroll
    for (int s = 0; s < 2; ++s) {
      const int c = tid + s * 256;
      const int row = c >> 3, col = (c & 7) * 8;
      async_copy_b128((uint32_t)(uintptr_t)&bsh[b][row][col],
                      Bt + (size_t)(n0 + row) * DMODEL + k0 + col);
    }
#pragma unroll
    for (int s = 0; s < 4; ++s) {
      const int c = tid + s * 256;
      const int row = c >> 3, col = (c & 7) * 8;
      async_copy_b128((uint32_t)(uintptr_t)&ash[b][row][col],
                      A + (size_t)(mblk + row) * DMODEL + k0 + col);
    }
  };

  prefetch(0, 0);
  v8f acc[4] = {};
  int buf = 0;
  for (int k0 = 0; k0 < DMODEL; k0 += 64) {
    if (k0 + 64 < DMODEL) {
      prefetch(buf ^ 1, k0 + 64);
      asm volatile("s_wait_asynccnt 0x6" ::: "memory");  // current tiles done, 6 newest in flight
    } else {
      asm volatile("s_wait_asynccnt 0x0" ::: "memory");
    }
    __syncthreads();

#pragma unroll
    for (int s = 0; s < 2; ++s) {
      const v16h a = load_frag_a(&ash[buf][wave * 16 + ln][0], s * 32, hi);
      v16h bf[4];
#pragma unroll
      for (int t = 0; t < 4; ++t) bf[t] = load_frag_b(&bsh[buf][t * 16 + ln][0], s * 32, hi);
#pragma unroll
      for (int t = 0; t < 4; ++t) acc[t] = wmma_f16(a, bf[t], acc[t]);
    }
    __syncthreads();
    buf ^= 1;
  }

#pragma unroll
  for (int t = 0; t < 4; ++t) {
    const int n = n0 + t * 16 + ln;
    const float bv = bias[n];
#pragma unroll
    for (int r = 0; r < 8; ++r) {
      const int m = m0 + r + hi * 8;
      const float val = acc[t][r] + bv;
      if (mode == 3) {
        out32[(size_t)m * DMODEL + n] = val;
      } else {
        const int h = n >> 6, dh = n & 63;
        if (mode == 2) out16[((size_t)h * HDIM + dh) * S_LEN + m] = (_Float16)val;
        else           out16[((size_t)h * S_LEN + m) * HDIM + dh] = (_Float16)val;
      }
    }
  }
}

// ---------------- sliding-window flash attention ----------------
// One block (4 waves) = 64 consecutive queries of one head. All waves walk the
// union key range; K/V chunk tiles are staged in LDS via async DMA, double
// buffered, and shared by the 4 waves. Out-of-window chunks are masked to
// -10000 and exactly annihilated by the online softmax.
__global__ __launch_bounds__(128) void attn_kernel(
    const _Float16* __restrict__ Qh,   // [H][S][64]
    const _Float16* __restrict__ Kh,   // [H][S][64]
    const _Float16* __restrict__ Vt,   // [H][64][S]
    const int* __restrict__ amask,     // [S]
    _Float16* __restrict__ Ch) {       // [S][1024] heads concatenated
  __shared__ __align__(16) _Float16 ksh[2][32][64];   // 2 x 4KB: keys x d
  __shared__ __align__(16) _Float16 vsh[2][64][32];   // 2 x 4KB: d x keys
  __shared__ __align__(16) _Float16 pbuf[4][16][32];  // 4KB: per-wave P tile

  const int tid  = threadIdx.x;
  const int lane = tid & 31;
  const int wave = tid >> 5;
  const int hi = lane >> 4;
  const int ln = lane & 15;
  const int head = blockIdx.x >> 6;          // 64 query-blocks per head
  const int qb = (blockIdx.x & 63) * 64;     // block's 64-query base
  const int q0 = qb + wave * 16;             // this wave's 16-query tile

  const _Float16* Qb = Qh + (size_t)head * S_LEN * HDIM;
  const _Float16* Kb = Kh + (size_t)head * S_LEN * HDIM;
  const _Float16* Vb = Vt + (size_t)head * HDIM * S_LEN;

  const v16h qf0 = load_frag_a(Qb + (size_t)(q0 + ln) * HDIM, 0, hi);
  const v16h qf1 = load_frag_a(Qb + (size_t)(q0 + ln) * HDIM, 32, hi);

  v8f acc[4] = {};
  float mrow[8], lrow[8];
#pragma unroll
  for (int r = 0; r < 8; ++r) { mrow[r] = -1e30f; lrow[r] = 0.f; }

  // block-uniform union key range
  int kstart = qb - WIN; if (kstart < 0) kstart = 0;
  int kend = qb + 64 + WIN; if (kend > S_LEN) kend = S_LEN;

  // DMA mapping per 32-key chunk: K tile 256 chunks, V tile 256 chunks; 2+2/thread
  auto prefetch = [&](int b, int k0) {
#pragma unroll
    for (int s = 0; s < 2; ++s) {
      const int c = tid + s * 128;
      const int krow = c >> 3, kcol = (c & 7) * 8;
      async_copy_b128((uint32_t)(uintptr_t)&ksh[b][krow][kcol],
                      Kb + (size_t)(k0 + krow) * HDIM + kcol);
      const int vrow = c >> 2, vcol = (c & 3) * 8;
      async_copy_b128((uint32_t)(uintptr_t)&vsh[b][vrow][vcol],
                      Vb + (size_t)vrow * S_LEN + k0 + vcol);
    }
  };

  prefetch(0, kstart);
  int buf = 0;
  for (int k0 = kstart; k0 < kend; k0 += 32) {
    if (k0 + 32 < kend) {
      prefetch(buf ^ 1, k0 + 32);
      asm volatile("s_wait_asynccnt 0x4" ::: "memory");  // current K/V done, 4 newest in flight
    } else {
      asm volatile("s_wait_asynccnt 0x0" ::: "memory");
    }
    __syncthreads();

    // scores: two 16x16 tiles over the 32 staged keys; batch the 4 K-fragments
    v16h kf[4];
    kf[0] = load_frag_b(&ksh[buf][ln][0], 0, hi);
    kf[1] = load_frag_b(&ksh[buf][ln][0], 32, hi);
    kf[2] = load_frag_b(&ksh[buf][16 + ln][0], 0, hi);
    kf[3] = load_frag_b(&ksh[buf][16 + ln][0], 32, hi);
    v8f s0 = {}, s1 = {};
    s0 = wmma_f16(qf0, kf[0], s0);
    s0 = wmma_f16(qf1, kf[1], s0);
    s1 = wmma_f16(qf0, kf[2], s1);
    s1 = wmma_f16(qf1, kf[3], s1);

    const int j0 = k0 + ln, j1 = k0 + 16 + ln;
    const bool am0 = amask[j0] != 0;
    const bool am1 = amask[j1] != 0;

    float p0[8], p1[8], corr[8];
#pragma unroll
    for (int r = 0; r < 8; ++r) {
      const int i = q0 + r + hi * 8;
      float v0 = s0[r] * 0.125f;   // 1/sqrt(64)
      float v1 = s1[r] * 0.125f;
      int d0 = i - j0; d0 = d0 < 0 ? -d0 : d0;
      int d1 = i - j1; d1 = d1 < 0 ? -d1 : d1;
      v0 = (d0 <= WIN && am0) ? v0 : -10000.0f;
      v1 = (d1 <= WIN && am1) ? v1 : -10000.0f;
      // row max across the 16 lanes holding this row (xor<16 stays in-half)
      float rm = fmaxf(v0, v1);
      rm = fmaxf(rm, __shfl_xor(rm, 1));
      rm = fmaxf(rm, __shfl_xor(rm, 2));
      rm = fmaxf(rm, __shfl_xor(rm, 4));
      rm = fmaxf(rm, __shfl_xor(rm, 8));
      const float mnew = fmaxf(mrow[r], rm);
      corr[r] = __expf(mrow[r] - mnew);
      mrow[r] = mnew;
      p0[r] = __expf(v0 - mnew);
      p1[r] = __expf(v1 - mnew);
      float rs = p0[r] + p1[r];
      rs += __shfl_xor(rs, 1);
      rs += __shfl_xor(rs, 2);
      rs += __shfl_xor(rs, 4);
      rs += __shfl_xor(rs, 8);
      lrow[r] = lrow[r] * corr[r] + rs;
    }
#pragma unroll
    for (int t = 0; t < 4; ++t)
#pragma unroll
      for (int r = 0; r < 8; ++r) acc[t][r] *= corr[r];

    // C-layout P -> A-fragment layout via wave-private LDS tile (DS in-order per wave)
#pragma unroll
    for (int r = 0; r < 8; ++r) {
      pbuf[wave][r + hi * 8][ln] = (_Float16)p0[r];
      pbuf[wave][r + hi * 8][16 + ln] = (_Float16)p1[r];
    }
    const v16h pf = load_frag_a(&pbuf[wave][ln][0], 0, hi);

    // ctx += P * V from the staged V tile; batch the 4 V-fragments
    v16h vf[4];
#pragma unroll
    for (int t = 0; t < 4; ++t) vf[t] = load_frag_b(&vsh[buf][t * 16 + ln][0], 0, hi);
#pragma unroll
    for (int t = 0; t < 4; ++t) acc[t] = wmma_f16(pf, vf[t], acc[t]);

    __syncthreads();
    buf ^= 1;
  }

#pragma unroll
  for (int t = 0; t < 4; ++t)
#pragma unroll
    for (int r = 0; r < 8; ++r) {
      const float val = acc[t][r] / lrow[r];
      const int srow = q0 + r + hi * 8;
      Ch[(size_t)srow * DMODEL + head * HDIM + t * 16 + ln] = (_Float16)val;
    }
}

// ---------------- launch ----------------

extern "C" void kernel_launch(void* const* d_in, const int* in_sizes, int n_in,
                              void* d_out, int out_size, void* d_ws, size_t ws_size,
                              hipStream_t stream) {
  const float* x   = (const float*)d_in[0];
  const int*   am  = (const int*)d_in[1];
  const float* Wq  = (const float*)d_in[2];
  const float* bq  = (const float*)d_in[3];
  const float* Wk  = (const float*)d_in[4];
  const float* bk  = (const float*)d_in[5];
  const float* Wv  = (const float*)d_in[6];
  const float* bv  = (const float*)d_in[7];
  const float* Wo  = (const float*)d_in[8];
  const float* bo  = (const float*)d_in[9];

  char* w = (char*)d_ws;
  const size_t MB = 1024 * 1024;
  _Float16* xh  = (_Float16*)(w);            //  8 MB: x in f16
  _Float16* Wqt = (_Float16*)(w +  8 * MB);  //  2 MB each: transposed f16 weights
  _Float16* Wkt = (_Float16*)(w + 10 * MB);
  _Float16* Wvt = (_Float16*)(w + 12 * MB);
  _Float16* Wot = (_Float16*)(w + 14 * MB);
  _Float16* Qh  = (_Float16*)(w + 16 * MB);  //  8 MB: [H][S][64]
  _Float16* Kh  = (_Float16*)(w + 24 * MB);  //  8 MB: [H][S][64]
  _Float16* Vt  = (_Float16*)(w + 32 * MB);  //  8 MB: [H][64][S]
  _Float16* Ch  = (_Float16*)(w + 40 * MB);  //  8 MB: ctx [S][1024]

  cvt_f32_to_f16<<<(S_LEN * DMODEL) / 256, 256, 0, stream>>>(x, xh, S_LEN * DMODEL);
  cvt_transpose<<<(DMODEL * DMODEL) / 256, 256, 0, stream>>>(Wq, Wqt);
  cvt_transpose<<<(DMODEL * DMODEL) / 256, 256, 0, stream>>>(Wk, Wkt);
  cvt_transpose<<<(DMODEL * DMODEL) / 256, 256, 0, stream>>>(Wv, Wvt);
  cvt_transpose<<<(DMODEL * DMODEL) / 256, 256, 0, stream>>>(Wo, Wot);

  dim3 ggrid(S_LEN / 128, DMODEL / 64);  // 32 x 16
  wmma_gemm_kernel<<<ggrid, 256, 0, stream>>>(xh, Wqt, bq, Qh, nullptr, 0);
  wmma_gemm_kernel<<<ggrid, 256, 0, stream>>>(xh, Wkt, bk, Kh, nullptr, 1);
  wmma_gemm_kernel<<<ggrid, 256, 0, stream>>>(xh, Wvt, bv, Vt, nullptr, 2);

  attn_kernel<<<NHEAD * (S_LEN / 64), 128, 0, stream>>>(Qh, Kh, Vt, am, Ch);

  wmma_gemm_kernel<<<ggrid, 256, 0, stream>>>(Ch, Wot, bo, nullptr, (float*)d_out, 3);
}
